// Qwen3VLTextAttention_33809982554758
// MI455X (gfx1250) — matrix operand order repaired and checked by reading, not compiled, not run on hardware
//
#include <hip/hip_runtime.h>

// ---------------- problem constants ----------------
static constexpr int S   = 2048;
static constexpr int HID = 4096;
static constexpr int H   = 32;
static constexpr int KV  = 8;
static constexpr int D   = 128;
static constexpr int G   = H / KV;          // 4
static constexpr float EPS   = 1e-6f;
static constexpr float SCALE = 0.08838834764831845f;  // 1/sqrt(128)

// ---------------- vector types ----------------
typedef __attribute__((ext_vector_type(16))) __bf16 bf16x16;
typedef __attribute__((ext_vector_type(8)))  __bf16 bf16x8;
typedef __attribute__((ext_vector_type(4)))  __bf16 bf16x4;
typedef __attribute__((ext_vector_type(8)))  float  floatx8;

// ---------------- WMMA wrapper ----------------
__device__ inline floatx8 wmma_bf16(bf16x16 a, bf16x16 b, floatx8 c) {
  return __builtin_amdgcn_wmma_f32_16x16x32_bf16(false, a, false, b, (short)0, c,
                                                 false, false);
}

__device__ inline bf16x16 combine16(bf16x8 lo, bf16x8 hi) {
  return __builtin_shufflevector(lo, hi, 0, 1, 2, 3, 4, 5, 6, 7,
                                 8, 9, 10, 11, 12, 13, 14, 15);
}

// A fragment: 16x32 bf16 tile, row-major (k contiguous), row stride ld.
__device__ inline bf16x16 load_afrag(const __bf16* p, int ld) {
  const int lane = threadIdx.x & 31;
  const __bf16* r = p + (lane & 15) * ld + ((lane >> 4) * 8);
  bf16x8 lo = *reinterpret_cast<const bf16x8*>(r);
  bf16x8 hi = *reinterpret_cast<const bf16x8*>(r + 16);
  return combine16(lo, hi);
}

// B fragment from an N-major tile (p -> [n][k], k contiguous, row stride ld).
__device__ inline bf16x16 load_bfrag_k(const __bf16* p, int ld) {
  const int lane = threadIdx.x & 31;
  const __bf16* r = p + (lane & 15) * ld + ((lane >> 4) * 8);
  bf16x8 lo = *reinterpret_cast<const bf16x8*>(r);
  bf16x8 hi = *reinterpret_cast<const bf16x8*>(r + 16);
  return combine16(lo, hi);
}

// CDNA5 async global->LDS copy of one 16-byte chunk per lane (ASYNCcnt path).
__device__ inline void async_copy_b128(void* lds_dst, const void* gsrc) {
  unsigned lds_off = (unsigned)(uintptr_t)lds_dst;  // low 32 bits = LDS offset
  asm volatile("global_load_async_to_lds_b128 %0, %1, off"
               :: "v"(lds_off), "v"(gsrc) : "memory");
}
__device__ inline void wait_async0() {
  asm volatile("s_wait_asynccnt 0x0" ::: "memory");
}

// ---------------- fp32 -> bf16 conversion (row-major copy) ----------------
__global__ __launch_bounds__(256) void cvt_f32_to_bf16(const float* __restrict__ src,
                                                       __bf16* __restrict__ dst, int n4) {
  int i = blockIdx.x * blockDim.x + threadIdx.x;
  if (i < n4) {
    float4 v = reinterpret_cast<const float4*>(src)[i];
    bf16x4 o = {(__bf16)v.x, (__bf16)v.y, (__bf16)v.z, (__bf16)v.w};
    reinterpret_cast<bf16x4*>(dst)[i] = o;
  }
}

// ---------------- fp32 [K][N] -> bf16 [N][K] tiled transpose ----------------
__global__ __launch_bounds__(256) void cvt_transpose_bf16(const float* __restrict__ src,
                                                          __bf16* __restrict__ dst,
                                                          int K, int N) {
  __shared__ float tile[32][33];
  const int kb = blockIdx.y * 32, nb = blockIdx.x * 32;
  const int tx = threadIdx.x & 31, ty = threadIdx.x >> 5;  // 8 rows per pass
#pragma unroll
  for (int i = 0; i < 32; i += 8)
    tile[ty + i][tx] = src[(size_t)(kb + ty + i) * N + nb + tx];
  __syncthreads();
  const int n = threadIdx.x >> 3;          // 0..31
  const int k4 = (threadIdx.x & 7) * 4;    // 0..28
  bf16x4 o = {(__bf16)tile[k4 + 0][n], (__bf16)tile[k4 + 1][n],
              (__bf16)tile[k4 + 2][n], (__bf16)tile[k4 + 3][n]};
  *reinterpret_cast<bf16x4*>(dst + (size_t)(nb + n) * K + kb + k4) = o;
}

// ---------------- bf16 WMMA GEMM: C(f32, MxN) = A(MxK) x Bt(NxK)^T ----------------
// 128x128 tile / WG, 8 waves; double-buffered LDS fed by async global->LDS.
#define TK 64
__global__ __launch_bounds__(256) void gemm_bf16_wmma(const __bf16* __restrict__ A,
                                                      const __bf16* __restrict__ Bt,
                                                      float* __restrict__ C,
                                                      int M, int N, int K) {
  __shared__ __bf16 La[2][128][TK + 8];   // 2 x 18 KB
  __shared__ __bf16 Lb[2][128][TK + 8];   // 2 x 18 KB (N-major: [n][k])
  const int tid  = threadIdx.x;
  const int wave = tid >> 5;
  const int lane = tid & 31;
  const int mblk = blockIdx.y * 128;
  const int nblk = blockIdx.x * 128;
  const int wm = (wave & 3) * 32;   // 4 waves along M
  const int wn = (wave >> 2) * 64;  // 2 waves along N

  int rr[4], cc[4];
#pragma unroll
  for (int i = 0; i < 4; ++i) {
    int ch = i * 256 + tid;
    rr[i] = ch >> 3;
    cc[i] = (ch & 7) * 8;
  }

  floatx8 acc[2][4];
#pragma unroll
  for (int i = 0; i < 2; ++i)
#pragma unroll
    for (int j = 0; j < 4; ++j)
#pragma unroll
      for (int v = 0; v < 8; ++v) acc[i][j][v] = 0.0f;

  // Async tile fetch: straight 16B-chunk copies, no VGPR round-trip.
  auto issue_tile = [&](int buf, int kk) {
#pragma unroll
    for (int i = 0; i < 4; ++i) {
      async_copy_b128(&La[buf][rr[i]][cc[i]],
                      A + (size_t)(mblk + rr[i]) * K + kk + cc[i]);
      async_copy_b128(&Lb[buf][rr[i]][cc[i]],
                      Bt + (size_t)(nblk + rr[i]) * K + kk + cc[i]);
    }
  };

  const int nk = K / TK;
  issue_tile(0, 0);
  wait_async0();
  __syncthreads();

  for (int t = 0; t < nk; ++t) {
    const int cur = t & 1, nxt = cur ^ 1;
    if (t + 1 < nk) issue_tile(nxt, (t + 1) * TK);  // async engine fills back buffer
    if (t + 2 < nk) {                               // prefetch t+2 slab into GL2
      __builtin_prefetch(A + (size_t)(mblk + (tid >> 1)) * K + (t + 2) * TK + (tid & 1) * 32, 0, 1);
      __builtin_prefetch(Bt + (size_t)(nblk + (tid >> 1)) * K + (t + 2) * TK + (tid & 1) * 32, 0, 1);
    }
#pragma unroll
    for (int ks = 0; ks < 2; ++ks) {
      bf16x16 af[2], bfv[4];
      af[0] = load_afrag(&La[cur][wm][ks * 32], TK + 8);
      af[1] = load_afrag(&La[cur][wm + 16][ks * 32], TK + 8);
#pragma unroll
      for (int j = 0; j < 4; ++j) bfv[j] = load_bfrag_k(&Lb[cur][wn + j * 16][ks * 32], TK + 8);
#pragma unroll
      for (int i = 0; i < 2; ++i)
#pragma unroll
        for (int j = 0; j < 4; ++j) acc[i][j] = wmma_bf16(af[i], bfv[j], acc[i][j]);
    }
    if (t + 1 < nk) wait_async0();                  // lands after the WMMA block
    __syncthreads();
  }

  const int cn = lane & 15, chf = lane >> 4;
#pragma unroll
  for (int i = 0; i < 2; ++i)
#pragma unroll
    for (int j = 0; j < 4; ++j)
#pragma unroll
      for (int v = 0; v < 8; ++v) {
        int row = mblk + wm + i * 16 + v + 8 * chf;
        int col = nblk + wn + j * 16 + cn;
        C[(size_t)row * N + col] = acc[i][j][v];
      }
}

// ---------------- RMSNorm + RoPE + head-transpose (wave-per-row) ----------------
__global__ __launch_bounds__(128) void normrope_kernel(
    const float* __restrict__ Qf, const float* __restrict__ Kf, const float* __restrict__ Vf,
    const float* __restrict__ cosb, const float* __restrict__ sinb,
    const float* __restrict__ qnw, const float* __restrict__ knw,
    __bf16* __restrict__ Qb, __bf16* __restrict__ Kb, __bf16* __restrict__ Vb) {
  const int wave = threadIdx.x >> 5;
  const int lane = threadIdx.x & 31;
  const long row = (long)blockIdx.x * 4 + wave;
  const long QROWS = (long)H * S;
  const long KROWS = (long)KV * S;

  const float* src;
  __bf16* dst;
  const float* w = nullptr;
  int s;
  if (row < QROWS) {
    int hh = (int)(row / S); s = (int)(row % S);
    src = Qf + ((size_t)s * H + hh) * D;
    dst = Qb + ((size_t)hh * S + s) * D;
    w = qnw;
  } else if (row < QROWS + KROWS) {
    long r2 = row - QROWS;
    int hh = (int)(r2 / S); s = (int)(r2 % S);
    src = Kf + ((size_t)s * KV + hh) * D;
    dst = Kb + ((size_t)hh * S + s) * D;
    w = knw;
  } else {
    long r2 = row - QROWS - KROWS;
    int hh = (int)(r2 / S); s = (int)(r2 % S);
    src = Vf + ((size_t)s * KV + hh) * D;
    dst = Vb + ((size_t)hh * S + s) * D;
  }

  const int d0 = lane * 4;
  float4 xv = *reinterpret_cast<const float4*>(src + d0);
  float x[4] = {xv.x, xv.y, xv.z, xv.w};

  if (w) {
    float ss = x[0] * x[0] + x[1] * x[1] + x[2] * x[2] + x[3] * x[3];
#pragma unroll
    for (int o = 16; o > 0; o >>= 1) ss += __shfl_xor(ss, o, 32);
    float inv = rsqrtf(ss * (1.0f / D) + EPS);

    float4 cv = *reinterpret_cast<const float4*>(cosb + (size_t)s * D + d0);
    float4 sv = *reinterpret_cast<const float4*>(sinb + (size_t)s * D + d0);
    float c4[4] = {cv.x, cv.y, cv.z, cv.w};
    float s4[4] = {sv.x, sv.y, sv.z, sv.w};

    float xn[4], part[4];
#pragma unroll
    for (int j = 0; j < 4; ++j) xn[j] = x[j] * inv * w[d0 + j];
#pragma unroll
    for (int j = 0; j < 4; ++j) part[j] = __shfl_xor(xn[j], 16, 32);  // d ^ 64
#pragma unroll
    for (int j = 0; j < 4; ++j) {
      float rot = (d0 < 64) ? -part[j] : part[j];
      x[j] = xn[j] * c4[j] + rot * s4[j];
    }
  }
  bf16x4 o = {(__bf16)x[0], (__bf16)x[1], (__bf16)x[2], (__bf16)x[3]};
  *reinterpret_cast<bf16x4*>(dst + d0) = o;
}

// ---------------- flash attention (WMMA, online softmax, double-buffered) ----------------
// Block: 128 threads (4 waves) x 64 query rows of one head; keys tiled by 64.
// K tiles: async global->LDS; V tiles: registers -> transposed LDS. Ping-pong buffers.
__global__ __launch_bounds__(128) void attn_wmma(const __bf16* __restrict__ Qb,
                                                 const __bf16* __restrict__ Kb,
                                                 const __bf16* __restrict__ Vb,
                                                 const float* __restrict__ mask,
                                                 __bf16* __restrict__ Ob) {
  __shared__ __bf16 Lk[2][64][128 + 8];   // K tiles row-major [key][d] (buf0 = Q stage)
  __shared__ __bf16 LvT[2][128][64 + 8];  // V tiles transposed [d][key]
  __shared__ __bf16 Lp[4][16][64 + 8];    // P staging per wave

  const int tid = threadIdx.x, wave = tid >> 5, lane = tid & 31;
  const int h  = blockIdx.y;
  const int kv = h / G;
  const int qbase = blockIdx.x * 64;

  const __bf16* Qh = Qb + ((size_t)h * S + qbase) * D;
  const __bf16* Kh = Kb + (size_t)kv * S * D;
  const __bf16* Vh = Vb + (size_t)kv * S * D;

  // Stage Q strip 64x128 into Lk[0] asynchronously, build per-wave A fragments.
#pragma unroll
  for (int ii = 0; ii < 8; ++ii) {
    int i = ii * 128 + tid;
    int r = i >> 4, c = (i & 15) * 8;
    async_copy_b128(&Lk[0][r][c], Qh + (size_t)r * D + c);
  }
  wait_async0();
  __syncthreads();
  bf16x16 qf[4];
#pragma unroll
  for (int ks = 0; ks < 4; ++ks) qf[ks] = load_afrag(&Lk[0][wave * 16][ks * 32], 128 + 8);
  __syncthreads();

  float mrow[8], lrow[8];
  floatx8 oacc[8];
#pragma unroll
  for (int v = 0; v < 8; ++v) { mrow[v] = -1e30f; lrow[v] = 0.0f; }
#pragma unroll
  for (int t = 0; t < 8; ++t)
#pragma unroll
    for (int v = 0; v < 8; ++v) oacc[t][v] = 0.0f;

  const int cn = lane & 15, chf = lane >> 4;

  bf16x8 rv[8];
  auto issue_kv = [&](int buf, int kt) {
#pragma unroll
    for (int ii = 0; ii < 8; ++ii) {
      int i = ii * 128 + tid;
      int r = i >> 4, c = (i & 15) * 8;
      async_copy_b128(&Lk[buf][r][c], Kh + (size_t)(kt + r) * D + c);
      rv[ii] = *reinterpret_cast<const bf16x8*>(Vh + (size_t)(kt + r) * D + c);
    }
  };
  auto store_v = [&](int buf) {
#pragma unroll
    for (int ii = 0; ii < 8; ++ii) {
      int i = ii * 128 + tid;
      int r = i >> 4, c = (i & 15) * 8;
#pragma unroll
      for (int j = 0; j < 8; ++j) LvT[buf][c + j][r] = rv[ii][j];
    }
  };

  // Prologue: tile 0 into buffer 0.
  issue_kv(0, 0);
  store_v(0);
  wait_async0();
  __syncthreads();

  const int NT = S / 64;
  for (int t = 0; t < NT; ++t) {
    const int cur = t & 1, nxt = cur ^ 1;
    const int kt = t * 64;
    if (t + 1 < NT) issue_kv(nxt, kt + 64);  // next K async + next V into registers

    // S = Q K^T : wave computes 16 (rows) x 64 (keys) = 4 C tiles.
    floatx8 sc[4];
#pragma unroll
    for (int j = 0; j < 4; ++j) {
      floatx8 a;
#pragma unroll
      for (int v = 0; v < 8; ++v) a[v] = 0.0f;
#pragma unroll
      for (int ks = 0; ks < 4; ++ks) {
        bf16x16 bfr = load_bfrag_k(&Lk[cur][j * 16][ks * 32], 128 + 8);
        a = wmma_bf16(qf[ks], bfr, a);
      }
#pragma unroll
      for (int v = 0; v < 8; ++v) {
        int qrow = qbase + wave * 16 + v + 8 * chf;
        int kcol = kt + j * 16 + cn;
        a[v] = a[v] * SCALE + mask[(size_t)qrow * S + kcol];
      }
      sc[j] = a;
    }

    // Online softmax: row v+8*half lives in one 16-lane half -> xor {8,4,2,1}.
#pragma unroll
    for (int v = 0; v < 8; ++v) {
      float mx = sc[0][v];
#pragma unroll
      for (int j = 1; j < 4; ++j) mx = fmaxf(mx, sc[j][v]);
#pragma unroll
      for (int o = 8; o > 0; o >>= 1) mx = fmaxf(mx, __shfl_xor(mx, o, 32));
      float mnew = fmaxf(mrow[v], mx);
      float corr = __expf(mrow[v] - mnew);
      float rsum = 0.0f;
#pragma unroll
      for (int j = 0; j < 4; ++j) {
        float p = __expf(sc[j][v] - mnew);
        sc[j][v] = p;
        rsum += p;
      }
#pragma unroll
      for (int o = 8; o > 0; o >>= 1) rsum += __shfl_xor(rsum, o, 32);
      lrow[v] = lrow[v] * corr + rsum;
      mrow[v] = mnew;
#pragma unroll
      for (int t2 = 0; t2 < 8; ++t2) oacc[t2][v] *= corr;
    }

    // Re-shape P: C layout -> LDS -> A fragments.
#pragma unroll
    for (int j = 0; j < 4; ++j)
#pragma unroll
      for (int v = 0; v < 8; ++v)
        Lp[wave][v + 8 * chf][j * 16 + cn] = (__bf16)sc[j][v];
    __syncthreads();

    // O += P V : kdim 64 -> 2 WMMA k-steps; 8 output d-tiles.
#pragma unroll
    for (int ks = 0; ks < 2; ++ks) {
      bf16x16 pf = load_afrag(&Lp[wave][0][ks * 32], 64 + 8);
#pragma unroll
      for (int t2 = 0; t2 < 8; ++t2) {
        bf16x16 vf = load_bfrag_k(&LvT[cur][t2 * 16][ks * 32], 64 + 8);
        oacc[t2] = wmma_bf16(pf, vf, oacc[t2]);
      }
    }

    if (t + 1 < NT) {
      store_v(nxt);    // loadcnt wait lands here, after all the compute
      wait_async0();   // next K tile resident in LDS
    }
    __syncthreads();
  }

  // Epilogue: one reciprocal per row, then scale+store bf16.
  float rl[8];
#pragma unroll
  for (int v = 0; v < 8; ++v) rl[v] = 1.0f / lrow[v];
#pragma unroll
  for (int t = 0; t < 8; ++t)
#pragma unroll
    for (int v = 0; v < 8; ++v) {
      int qrow = qbase + wave * 16 + v + 8 * chf;
      float val = oacc[t][v] * rl[v];
      Ob[(size_t)qrow * (H * D) + (size_t)h * D + t * 16 + cn] = (__bf16)val;
    }
}

// ---------------- launcher ----------------
extern "C" void kernel_launch(void* const* d_in, const int* in_sizes, int n_in,
                              void* d_out, int out_size, void* d_ws, size_t ws_size,
                              hipStream_t stream) {
  (void)in_sizes; (void)n_in; (void)out_size; (void)ws_size;
  const float* hidden = (const float*)d_in[0];
  const float* cosb   = (const float*)d_in[1];
  const float* sinb   = (const float*)d_in[2];
  const float* mask   = (const float*)d_in[3];
  const float* wq     = (const float*)d_in[4];
  const float* wk     = (const float*)d_in[5];
  const float* wv     = (const float*)d_in[6];
  const float* wo     = (const float*)d_in[7];
  const float* qnw    = (const float*)d_in[8];
  const float* knw    = (const float*)d_in[9];
  float* out = (float*)d_out;

  char* ws = (char*)d_ws;
  size_t off = 0;
  auto carve = [&](size_t bytes) -> char* {
    char* p = ws + off;
    off += (bytes + 255) & ~(size_t)255;
    return p;
  };
  __bf16* Xb  = (__bf16*)carve((size_t)S * HID * 2);
  __bf16* Wqt = (__bf16*)carve((size_t)HID * HID * 2);      // [N=4096][K=4096]
  __bf16* Wkt = (__bf16*)carve((size_t)HID * KV * D * 2);   // [N=1024][K=4096]
  __bf16* Wvt = (__bf16*)carve((size_t)HID * KV * D * 2);
  __bf16* Wot = (__bf16*)carve((size_t)HID * HID * 2);
  float*  Qf  = (float*)carve((size_t)S * HID * 4);
  float*  Kf  = (float*)carve((size_t)S * KV * D * 4);
  float*  Vf  = (float*)carve((size_t)S * KV * D * 4);
  __bf16* Qb  = (__bf16*)carve((size_t)H * S * D * 2);
  __bf16* Kb  = (__bf16*)carve((size_t)KV * S * D * 2);
  __bf16* Vb  = (__bf16*)carve((size_t)KV * S * D * 2);
  __bf16* Ab  = (__bf16*)carve((size_t)S * HID * 2);

  {
    int n4 = (int)((size_t)S * HID / 4);
    cvt_f32_to_bf16<<<dim3((n4 + 255) / 256), dim3(256), 0, stream>>>(hidden, Xb, n4);
  }
  cvt_transpose_bf16<<<dim3(HID / 32, HID / 32), dim3(256), 0, stream>>>(wq, Wqt, HID, HID);
  cvt_transpose_bf16<<<dim3((KV * D) / 32, HID / 32), dim3(256), 0, stream>>>(wk, Wkt, HID, KV * D);
  cvt_transpose_bf16<<<dim3((KV * D) / 32, HID / 32), dim3(256), 0, stream>>>(wv, Wvt, HID, KV * D);
  cvt_transpose_bf16<<<dim3(HID / 32, HID / 32), dim3(256), 0, stream>>>(wo, Wot, HID, HID);

  gemm_bf16_wmma<<<dim3(HID / 128, S / 128), dim3(256), 0, stream>>>(Xb, Wqt, Qf, S, HID, HID);
  gemm_bf16_wmma<<<dim3((KV * D) / 128, S / 128), dim3(256), 0, stream>>>(Xb, Wkt, Kf, S, KV * D, HID);
  gemm_bf16_wmma<<<dim3((KV * D) / 128, S / 128), dim3(256), 0, stream>>>(Xb, Wvt, Vf, S, KV * D, HID);

  normrope_kernel<<<dim3((S * (H + 2 * KV)) / 4), dim3(128), 0, stream>>>(
      Qf, Kf, Vf, cosb, sinb, qnw, knw, Qb, Kb, Vb);

  attn_wmma<<<dim3(S / 64, H), dim3(128), 0, stream>>>(Qb, Kb, Vb, mask, Ab);

  gemm_bf16_wmma<<<dim3(HID / 128, S / 128), dim3(256), 0, stream>>>(Ab, Wot, out, S, HID, HID);
}
